// DeepseekV3GroupedExperts_5669356830747
// MI455X (gfx1250) — matrix operand order repaired and checked by reading, not compiled, not run on hardware
//
#include <hip/hip_runtime.h>

typedef __attribute__((ext_vector_type(16))) __bf16 v16bf;
typedef __attribute__((ext_vector_type(8)))  __bf16 v8bf;
typedef __attribute__((ext_vector_type(8)))  float  v8f;
typedef unsigned int u32x4 __attribute__((ext_vector_type(4)));
typedef int          i32x8 __attribute__((ext_vector_type(8)));
typedef int          i32x4 __attribute__((ext_vector_type(4)));

#define DEVFN static __device__ __forceinline__

constexpr int NE   = 8;
constexpr int DIM  = 2048;
constexpr int HID  = 1408;
constexpr int TPE  = 4096;
constexpr int TOK  = NE * TPE;

// ---------------- helpers ----------------
DEVFN unsigned short f32_to_bf16(float f) {
    unsigned u = __builtin_bit_cast(unsigned, f);
    unsigned r = u + 0x7FFFu + ((u >> 16) & 1u);   // round-to-nearest-even
    return (unsigned short)(r >> 16);
}

DEVFN v16bf cat16(v8bf lo, v8bf hi) {
    return __builtin_shufflevector(lo, hi, 0,1,2,3,4,5,6,7,8,9,10,11,12,13,14,15);
}

// Wave-relative LDS byte address of a generic pointer into __shared__.
DEVFN unsigned lds_addr_of(const void* p) {
    return (unsigned)(unsigned long long)
        (__attribute__((address_space(3))) const void*)p;
}

// CDNA5 per-lane async copy: 16 bytes global -> LDS, tracked by ASYNCcnt.
DEVFN void async_copy16(unsigned lds_byte, const void* gptr) {
    asm volatile("global_load_async_to_lds_b128 %0, %1, off"
                 :: "v"(lds_byte), "v"(gptr) : "memory");
}

DEVFN void wait_async0() {
    asm volatile("s_wait_asynccnt 0x0" ::: "memory");
}

// CDNA5 Tensor Data Mover: DMA a 2-D tile (tile_k x rows, bf16) from global
// memory (row stride `row_stride` elements) into LDS at byte offset lds_byte.
// D# packing per CDNA5 ISA. One issue per wave;
// completion tracked by TENSORcnt. This toolchain's builtin takes 6 args:
// (g0 v4u, g1 v8i, g2 v4i, g3 v4i, v8i, cpol) -- trailing groups unused for 2-D.
DEVFN void tdm_load_2d(unsigned lds_byte, const void* gaddr,
                       unsigned tile_k, unsigned rows, unsigned row_stride) {
    const unsigned long long ga = (unsigned long long)gaddr;
    u32x4 g0;
    g0[0] = 1u;                                        // count=1, user mode
    g0[1] = lds_byte;                                  // lds_addr
    g0[2] = (unsigned)ga;                              // global_addr[31:0]
    g0[3] = (unsigned)((ga >> 32) & 0x01FFFFFFu)       // global_addr[56:32]
            | 0x80000000u;                             // type=2 ("image")
    i32x8 g1;
    g1[0] = 0x00010000;                                // data_size=1 (2 bytes)
    g1[1] = (int)((row_stride & 0xFFFFu) << 16);       // tensor_dim0[15:0]
    g1[2] = (int)(((row_stride >> 16) & 0xFFFFu)       // tensor_dim0[31:16]
            | ((rows & 0xFFFFu) << 16));               // tensor_dim1[15:0]
    g1[3] = (int)(((rows >> 16) & 0xFFFFu)             // tensor_dim1[31:16]
            | ((tile_k & 0xFFFFu) << 16));             // tile_dim0
    g1[4] = (int)(rows & 0xFFFFu);                     // tile_dim1 (tile_dim2=0)
    g1[5] = (int)row_stride;                           // tensor_dim0_stride[31:0]
    g1[6] = 0;                                         // stride0 hi | stride1 lo
    g1[7] = 0;                                         // stride1 hi
    const i32x4 z4 = {0, 0, 0, 0};                     // 2-D: groups 2/3 unused
    const i32x8 z8 = {0, 0, 0, 0, 0, 0, 0, 0};
    __builtin_amdgcn_tensor_load_to_lds(g0, g1, z4, z4, z8, 0);
}

// Load A fragment (16x32 bf16, rows m0..m0+15) from LDS tile laid out [row][32].
// Lane L (L<16): row L, K = {0..7, 16..23}; lane L>=16: row L-16, K = {8..15, 24..31}.
DEVFN v16bf load_a_frag(const unsigned short* lds, int m0, int lane) {
    const int lrow = lane & 15;
    const int lkb  = (lane >> 4) * 8;
    const unsigned short* base = lds + (m0 + lrow) * 32 + lkb;
    v8bf lo = *(const v8bf*)(base);
    v8bf hi = *(const v8bf*)(base + 16);
    return cat16(lo, hi);
}

// Load B fragment (32x16 bf16, cols n0..n0+15) from LDS tile laid out [col][32].
// Lane L<16: col = L%16, K = {0..7, 16..23}; lane L>=16: K = {8..15, 24..31}.
DEVFN v16bf load_b_frag(const unsigned short* lds, int n0, int lane) {
    const int lcol = lane & 15;
    const int lkb  = (lane >> 4) * 8;
    const unsigned short* base = lds + (n0 + lcol) * 32 + lkb;
    v8bf lo = *(const v8bf*)(base);
    v8bf hi = *(const v8bf*)(base + 16);
    return cat16(lo, hi);
}

// ---------------- stage 1: x f32 -> bf16 ----------------
__global__ __launch_bounds__(256)
void cvt_x_kernel(const float* __restrict__ src, unsigned short* __restrict__ dst, int n) {
    int i = (blockIdx.x * 256 + threadIdx.x) * 4;
    if (i + 3 < n) {
        float4 v = *(const float4*)(src + i);
        uint2 o;
        o.x = (unsigned)f32_to_bf16(v.x) | ((unsigned)f32_to_bf16(v.y) << 16);
        o.y = (unsigned)f32_to_bf16(v.z) | ((unsigned)f32_to_bf16(v.w) << 16);
        *(uint2*)(dst + i) = o;
    }
}

// ---------------- stage 2: weight f32 [E][R][C] -> bf16 transposed [E][C][R] ----
__global__ __launch_bounds__(256)
void cvt_tr_kernel(const float* __restrict__ src, unsigned short* __restrict__ dst,
                   int R, int C) {
    __shared__ float tile[32][33];
    const int e  = blockIdx.z;
    const int c0 = blockIdx.x * 32;
    const int r0 = blockIdx.y * 32;
    const float*    s = src + (size_t)e * R * C;
    unsigned short* d = dst + (size_t)e * R * C;
    const int tx = threadIdx.x & 31;
    const int ty = threadIdx.x >> 5;
    for (int rr = ty; rr < 32; rr += 8)
        tile[rr][tx] = s[(size_t)(r0 + rr) * C + c0 + tx];
    __syncthreads();
    for (int cc = ty; cc < 32; cc += 8)
        d[(size_t)(c0 + cc) * R + r0 + tx] = f32_to_bf16(tile[tx][cc]);
}

// ---------------- stage 3: fused gate+up GEMM + SwiGLU (TDM staging) ---------
// A = xb [TOK][DIM] bf16 ; B = wgT/wuT [E][HID][DIM] bf16 (K-contiguous)
// Wave 0 issues 3 TDM tile loads for the next LDS buffer; all waves compute
// WMMAs on the current buffer; wave 0 waits TENSORcnt; one barrier per K-step.
// Dyn LDS layout (bytes): A[2]@0 (2*8192) | Bg[2]@16384 (2*4096) | Bu[2]@24576 (2*4096)
__global__ __launch_bounds__(256)
void moe_gateup_kernel(const unsigned short* __restrict__ xb,
                       const unsigned short* __restrict__ wgT,
                       const unsigned short* __restrict__ wuT,
                       unsigned short* __restrict__ hbuf) {
    extern __shared__ __align__(16) unsigned short smem[];

    const int e    = blockIdx.z;
    const int t0   = e * TPE + blockIdx.y * 128;
    const int n0   = blockIdx.x * 64;
    const int tid  = threadIdx.x;
    const int lane = tid & 31;
    const int wv   = tid >> 5;
    const int wm   = wv & 3;
    const int wn   = wv >> 2;

    const unsigned short* A  = xb  + (size_t)t0 * DIM;
    const unsigned short* Bg = wgT + ((size_t)e * HID + n0) * DIM;
    const unsigned short* Bu = wuT + ((size_t)e * HID + n0) * DIM;

    const unsigned lbase = lds_addr_of(smem);
    constexpr unsigned A_B = 128 * 32 * 2;    // 8192
    constexpr unsigned B_B = 64 * 32 * 2;     // 4096
    constexpr int NK = DIM / 32;              // 64

    auto stage = [&](int buf, int kb) {
        if (wv == 0) {
            tdm_load_2d(lbase + buf * A_B,                     A  + kb, 32, 128, DIM);
            tdm_load_2d(lbase + 2 * A_B + buf * B_B,           Bg + kb, 32,  64, DIM);
            tdm_load_2d(lbase + 2 * A_B + 2 * B_B + buf * B_B, Bu + kb, 32,  64, DIM);
        }
    };
    auto sync_tiles = [&]() {
        if (wv == 0) __builtin_amdgcn_s_wait_tensorcnt(0);
        __syncthreads();
    };

    v8f accg[2][2] = {};
    v8f accu[2][2] = {};

    stage(0, 0);
    sync_tiles();

    for (int kbi = 0; kbi < NK; ++kbi) {
        const int cur = kbi & 1;
        if (kbi + 1 < NK) stage(1 - cur, (kbi + 1) * 32);

        const unsigned short* sa  = smem + cur * (A_B / 2);
        const unsigned short* sbg = smem + (2 * A_B) / 2 + cur * (B_B / 2);
        const unsigned short* sbu = smem + (2 * A_B + 2 * B_B) / 2 + cur * (B_B / 2);

        v16bf af[2], bgf[2], buf2[2];
        #pragma unroll
        for (int i = 0; i < 2; ++i) af[i]   = load_a_frag(sa,  wm * 32 + i * 16, lane);
        #pragma unroll
        for (int j = 0; j < 2; ++j) bgf[j]  = load_b_frag(sbg, wn * 32 + j * 16, lane);
        #pragma unroll
        for (int j = 0; j < 2; ++j) buf2[j] = load_b_frag(sbu, wn * 32 + j * 16, lane);

        #pragma unroll
        for (int i = 0; i < 2; ++i)
            #pragma unroll
            for (int j = 0; j < 2; ++j) {
                accg[i][j] = __builtin_amdgcn_wmma_f32_16x16x32_bf16(
                    false, af[i], false, bgf[j], (short)0, accg[i][j], false, false);
                accu[i][j] = __builtin_amdgcn_wmma_f32_16x16x32_bf16(
                    false, af[i], false, buf2[j], (short)0, accu[i][j], false, false);
            }

        sync_tiles();
    }

    // epilogue: h = silu(g) * u  (C frag: vgpr r -> row (lane/16)*8 + r, col lane%16)
    const int mlo = (lane >> 4) * 8;
    const int nlo = lane & 15;
    #pragma unroll
    for (int i = 0; i < 2; ++i)
        #pragma unroll
        for (int j = 0; j < 2; ++j) {
            const int mbase = t0 + wm * 32 + i * 16 + mlo;
            const int ncol  = n0 + wn * 32 + j * 16 + nlo;
            #pragma unroll
            for (int r = 0; r < 8; ++r) {
                float g = accg[i][j][r];
                float u = accu[i][j][r];
                float s = g / (1.0f + __expf(-g));   // silu
                hbuf[(size_t)(mbase + r) * HID + ncol] = f32_to_bf16(s * u);
            }
        }
}

// ---------------- stage 4: down GEMM (per-lane async-LDS staging) -----------
// A = hbuf [TOK][HID] bf16 ; B = wdT [E][DIM][HID] bf16 ; out f32 [TOK][DIM]
// Dyn LDS layout (bytes): A[2]@0 (2*8192) | B[2]@16384 (2*4096)
__global__ __launch_bounds__(256)
void moe_down_kernel(const unsigned short* __restrict__ hbuf,
                     const unsigned short* __restrict__ wdT,
                     float* __restrict__ out) {
    extern __shared__ __align__(16) unsigned short smem[];

    const int e    = blockIdx.z;
    const int t0   = e * TPE + blockIdx.y * 128;
    const int n0   = blockIdx.x * 64;
    const int tid  = threadIdx.x;
    const int lane = tid & 31;
    const int wv   = tid >> 5;
    const int wm   = wv & 3;
    const int wn   = wv >> 2;

    const unsigned short* A = hbuf + (size_t)t0 * HID;
    const unsigned short* B = wdT  + ((size_t)e * DIM + n0) * HID;

    const unsigned lbase = lds_addr_of(smem);
    constexpr unsigned A_B = 128 * 32 * 2;
    constexpr unsigned B_B = 64 * 32 * 2;
    constexpr int NK = HID / 32;              // 44

    const int sA0 = tid, sA1 = tid + 256;
    const size_t gA0 = (size_t)(sA0 >> 2) * HID + (sA0 & 3) * 8;
    const size_t gA1 = (size_t)(sA1 >> 2) * HID + (sA1 & 3) * 8;
    const size_t gB  = (size_t)(tid >> 2) * HID + (tid & 3) * 8;

    auto stage = [&](int buf, int kb) {
        const unsigned aL = lbase + buf * A_B;
        const unsigned bL = lbase + 2 * A_B + buf * B_B;
        async_copy16(aL + (unsigned)sA0 * 16u, A + gA0 + kb);
        async_copy16(aL + (unsigned)sA1 * 16u, A + gA1 + kb);
        async_copy16(bL + (unsigned)tid * 16u, B + gB + kb);
    };

    v8f acc[2][2] = {};

    stage(0, 0);
    wait_async0();
    __syncthreads();

    for (int kbi = 0; kbi < NK; ++kbi) {
        const int cur = kbi & 1;
        if (kbi + 1 < NK) stage(1 - cur, (kbi + 1) * 32);

        const unsigned short* sa = smem + cur * (A_B / 2);
        const unsigned short* sb = smem + (2 * A_B) / 2 + cur * (B_B / 2);

        v16bf af[2], bf[2];
        #pragma unroll
        for (int i = 0; i < 2; ++i) af[i] = load_a_frag(sa, wm * 32 + i * 16, lane);
        #pragma unroll
        for (int j = 0; j < 2; ++j) bf[j] = load_b_frag(sb, wn * 32 + j * 16, lane);

        #pragma unroll
        for (int i = 0; i < 2; ++i)
            #pragma unroll
            for (int j = 0; j < 2; ++j)
                acc[i][j] = __builtin_amdgcn_wmma_f32_16x16x32_bf16(
                    false, af[i], false, bf[j], (short)0, acc[i][j], false, false);

        wait_async0();
        __syncthreads();
    }

    const int mlo = (lane >> 4) * 8;
    const int nlo = lane & 15;
    #pragma unroll
    for (int i = 0; i < 2; ++i)
        #pragma unroll
        for (int j = 0; j < 2; ++j) {
            const int mbase = t0 + wm * 32 + i * 16 + mlo;
            const int ncol  = n0 + wn * 32 + j * 16 + nlo;
            #pragma unroll
            for (int r = 0; r < 8; ++r)
                out[(size_t)(mbase + r) * DIM + ncol] = acc[i][j][r];
        }
}

// ---------------- host launcher ----------------
extern "C" void kernel_launch(void* const* d_in, const int* in_sizes, int n_in,
                              void* d_out, int out_size, void* d_ws, size_t ws_size,
                              hipStream_t stream) {
    const float* x      = (const float*)d_in[0];
    // d_in[1] = num_tokens_per_expert (static equal splits; unused)
    const float* w_gate = (const float*)d_in[2];
    const float* w_up   = (const float*)d_in[3];
    const float* w_down = (const float*)d_in[4];
    float*       out    = (float*)d_out;

    char* ws = (char*)d_ws;
    size_t off = 0;
    unsigned short* xb  = (unsigned short*)(ws + off); off += (size_t)TOK * DIM * 2;
    unsigned short* wgT = (unsigned short*)(ws + off); off += (size_t)NE * DIM * HID * 2;
    unsigned short* wuT = (unsigned short*)(ws + off); off += (size_t)NE * DIM * HID * 2;
    unsigned short* wdT = (unsigned short*)(ws + off); off += (size_t)NE * HID * DIM * 2;
    unsigned short* hb  = (unsigned short*)(ws + off); off += (size_t)TOK * HID * 2;

    // 1) x -> bf16
    {
        int n = TOK * DIM;
        cvt_x_kernel<<<n / (256 * 4), 256, 0, stream>>>(x, xb, n);
    }
    // 2) weights -> bf16, transposed to [N][K]
    cvt_tr_kernel<<<dim3(HID / 32, DIM / 32, NE), 256, 0, stream>>>(w_gate, wgT, DIM, HID);
    cvt_tr_kernel<<<dim3(HID / 32, DIM / 32, NE), 256, 0, stream>>>(w_up,   wuT, DIM, HID);
    cvt_tr_kernel<<<dim3(DIM / 32, HID / 32, NE), 256, 0, stream>>>(w_down, wdT, HID, DIM);
    // 3) fused gate/up + SwiGLU (TDM staging; dyn LDS 32 KB)
    moe_gateup_kernel<<<dim3(HID / 64, TPE / 128, NE), 256, 32768, stream>>>(xb, wgT, wuT, hb);
    // 4) down projection (async-LDS staging; dyn LDS 24 KB)
    moe_down_kernel<<<dim3(DIM / 64, TPE / 128, NE), 256, 24576, stream>>>(hb, wdT, out);

    (void)in_sizes; (void)n_in; (void)out_size; (void)ws_size; (void)off;
}
